// IIN_29162827940550
// MI455X (gfx1250) — matrix operand order repaired and checked by reading, not compiled, not run on hardware
//
#include <hip/hip_runtime.h>
#include <hip/hip_bf16.h>
#include <math.h>

// ---------------------------------------------------------------------------
// Types for CDNA5 WMMA (wave32): D(16x16 f32) = A(16x32 bf16) x B(32x16 bf16) + C
// ---------------------------------------------------------------------------
typedef __attribute__((ext_vector_type(16))) __bf16 v16bf;
typedef __attribute__((ext_vector_type(8)))  __bf16 v8bf;
typedef __attribute__((ext_vector_type(8)))  float  v8f;

#define H_DIM 256     // hidden size
#define K_DIM 512     // 2*H
#define TB    16      // batch rows per workgroup

// ---------------------------------------------------------------------------
// Fragment loaders (k = 16*grp + e in both A and B; B is prepacked so each
// lane's 16 bf16 are contiguous -> two global_load_b128 per fragment).
// ---------------------------------------------------------------------------
__device__ __forceinline__ v16bf load_A_lds(const __bf16* base, int ld, int kbase) {
  int lane = threadIdx.x & 31;
  int grp  = lane >> 4;
  int m    = lane & 15;
  const __bf16* row = base + m * ld + kbase;
  union { v16bf v; v8bf h[2]; } u;
  u.h[0] = *(const v8bf*)(row + grp * 16);
  u.h[1] = *(const v8bf*)(row + grp * 16 + 8);
  return u.v;
}

__device__ __forceinline__ v16bf load_B_pk(const __bf16* pk, int ntiles, int kt, int nt) {
  int lane = threadIdx.x & 31;
  const __bf16* p = pk + (((size_t)(kt * ntiles + nt)) * 32 + (size_t)lane) * 16;
  union { v16bf v; v8bf h[2]; } u;
  u.h[0] = *(const v8bf*)(p);
  u.h[1] = *(const v8bf*)(p + 8);
  return u.v;
}

// Two 16x16 n-tiles (nt0, nt0+1) over kTiles K-blocks.  One shared A stream
// (halves LDS traffic; each A load feeds 2 WMMAs), 1-deep software pipeline
// so next fragments' load latency overlaps the current WMMAs.  unroll(2)
// lets the register allocator coalesce the ping-pong rotation (a=an etc.)
// into pure renames instead of 16 v_mov_b32 per iteration (seen in round 3).
__device__ __forceinline__ void gemm_tile2(const __bf16* aBase, const __bf16* pk,
                                           int ntiles, int nt0, int kTiles,
                                           v8f& acc0, v8f& acc1) {
  v16bf a  = load_A_lds(aBase, K_DIM, 0);
  v16bf b0 = load_B_pk(pk, ntiles, 0, nt0);
  v16bf b1 = load_B_pk(pk, ntiles, 0, nt0 + 1);
  #pragma unroll 2
  for (int kt = 0; kt < kTiles - 1; ++kt) {
    v16bf an  = load_A_lds(aBase, K_DIM, (kt + 1) * 32);
    v16bf bn0 = load_B_pk(pk, ntiles, kt + 1, nt0);
    v16bf bn1 = load_B_pk(pk, ntiles, kt + 1, nt0 + 1);
    acc0 = __builtin_amdgcn_wmma_f32_16x16x32_bf16(
               false, a, false, b0, (short)0, acc0, false, false);
    acc1 = __builtin_amdgcn_wmma_f32_16x16x32_bf16(
               false, a, false, b1, (short)0, acc1, false, false);
    a = an; b0 = bn0; b1 = bn1;
  }
  acc0 = __builtin_amdgcn_wmma_f32_16x16x32_bf16(
             false, a, false, b0, (short)0, acc0, false, false);
  acc1 = __builtin_amdgcn_wmma_f32_16x16x32_bf16(
             false, a, false, b1, (short)0, acc1, false, false);
}

// Fast activations: avoid precise v_div_scale/v_rcp/v_fma IEEE division
// chains on the recurrence critical path; v_exp_f32 + v_rcp_f32 instead.
__device__ __forceinline__ float fast_sigmoid(float x) {
  return __fdividef(1.f, 1.f + __expf(-x));
}
__device__ __forceinline__ float fast_tanh(float x) {
  return 1.f - __fdividef(2.f, __expf(2.f * x) + 1.f);
}

// ---------------------------------------------------------------------------
// Kernel 1: attention logits + masked softmax.  One block per batch row.
// Pure streaming (200 MB of hist) -> memory bound, no WMMA needed.
// ---------------------------------------------------------------------------
__global__ __launch_bounds__(256) void attn_kernel(
    const float* __restrict__ hist, const float* __restrict__ target,
    const int* __restrict__ mask, float* __restrict__ att, int S)
{
  const int b   = blockIdx.x;
  const int tid = threadIdx.x;
  __shared__ float tgt[H_DIM];
  __shared__ float lg[256];
  __shared__ float red[256];

  tgt[tid] = target[(size_t)b * H_DIM + tid];
  __syncthreads();

  const int wave = tid >> 5, lane = tid & 31;
  for (int s = wave; s < S; s += 8) {
    const float* row = hist + ((size_t)b * S + s) * H_DIM;
    float acc = 0.f;
    for (int k = lane; k < H_DIM; k += 32) acc += row[k] * tgt[k];
    for (int off = 16; off > 0; off >>= 1) acc += __shfl_xor(acc, off, 32);
    if (lane == 0) {
      float l = acc * 0.0625f;                       // 1/sqrt(256)
      lg[s] = (mask[(size_t)b * S + s] == 0) ? -1.0e9f : l;
    }
  }
  __syncthreads();

  float v = (tid < S) ? lg[tid] : -1.0e30f;
  red[tid] = v; __syncthreads();
  for (int off = 128; off > 0; off >>= 1) {
    if (tid < off) red[tid] = fmaxf(red[tid], red[tid + off]);
    __syncthreads();
  }
  float mx = red[0]; __syncthreads();
  float e = (tid < S) ? __expf(lg[tid] - mx) : 0.f;
  red[tid] = e; __syncthreads();
  for (int off = 128; off > 0; off >>= 1) {
    if (tid < off) red[tid] += red[tid + off];
    __syncthreads();
  }
  float denom = red[0];
  if (tid < S) att[(size_t)b * S + tid] = __fdividef(e, denom);
}

// ---------------------------------------------------------------------------
// Kernel 2: prepack W (N x K fp32, row = output unit) into bf16 B-fragment
// tiles.  Tile(kt,nt) holds 32 lanes x 16 contiguous bf16:
//   lane l (grp=l>>4, n=l&15), elem e  <-  W[nt*16+n][kt*32+grp*16+e]
// ---------------------------------------------------------------------------
__global__ void prepack_kernel(const float* __restrict__ W, __bf16* __restrict__ pk,
                               int N, int K)
{
  int idx = blockIdx.x * blockDim.x + threadIdx.x;
  int ntiles = N / 16;
  int total  = (K / 32) * ntiles * 32;
  if (idx >= total) return;
  int lane = idx & 31;
  int tile = idx >> 5;
  int nt = tile % ntiles, kt = tile / ntiles;
  int grp = lane >> 4, n = lane & 15;
  const float* src = W + (size_t)(nt * 16 + n) * K + (kt * 32 + grp * 16);
  __bf16* dst = pk + ((size_t)tile * 32 + lane) * 16;
  #pragma unroll
  for (int e = 0; e < 16; ++e) dst[e] = (__bf16)src[e];
}

// ---------------------------------------------------------------------------
// Kernel 3: fused GRU-attention scan.  Each workgroup owns TB=16 batch rows
// and runs all S steps (recurrence independent per row -> no cross-WG sync).
// 4 waves / 128 threads.  Per step:
//   gates:     comb[x|h](16x512) @ [Wr^T,Wu^T](512x512)  -> waves 0-1: Wr,
//              waves 2-3: Wu (uniform per wave, no per-tile selects)
//   candidate: comb[x|r*h](16x512) @ Wc^T(512x256)
// h kept fp32 in LDS; WMMA operands bf16 with f32 accumulate.
// Pair loops stay ROLLED so t-invariant B fragments are not hoisted/spilled.
// ---------------------------------------------------------------------------
__global__ __launch_bounds__(128) void gru_fused_kernel(
    const float* __restrict__ hist, const int* __restrict__ mask,
    const float* __restrict__ att,
    const float* __restrict__ br, const float* __restrict__ bu,
    const float* __restrict__ bc, const float* __restrict__ bo,
    const __bf16* __restrict__ WrP, const __bf16* __restrict__ WuP,
    const __bf16* __restrict__ WcP, const __bf16* __restrict__ WoP,
    float* __restrict__ out, int S)
{
  __shared__ __bf16 comb[TB][K_DIM];   // [x | h]  (or [x | r*h])   16 KB
  __shared__ float  hbuf[TB][H_DIM];   // master hidden state fp32  16 KB
  __shared__ __bf16 gbuf[TB][K_DIM];   // r (0:256) and u (256:512) 16 KB
  __shared__ __bf16 mxh [TB][H_DIM];   // masked running max         8 KB
  __shared__ float  att_t[TB];
  __shared__ int    mask_t[TB];

  const int bBase = blockIdx.x * TB;
  const int tid   = threadIdx.x;
  const int wave  = tid >> 5;
  const int lane  = tid & 31;
  const int grp   = lane >> 4;
  const int nIdx  = lane & 15;

  for (int i = tid; i < TB * H_DIM; i += 128) {
    ((float*)hbuf)[i] = 0.f;
    ((__bf16*)mxh)[i] = (__bf16)(-1.0e30f);
    comb[i / H_DIM][H_DIM + (i % H_DIM)] = (__bf16)0.f;   // h0 = 0
  }
  __syncthreads();

  // uniform-per-wave gate assignment: waves 0-1 -> r (Wr), waves 2-3 -> u (Wu)
  const __bf16* WPg   = (wave < 2) ? WrP : WuP;
  const float*  biasg = (wave < 2) ? br  : bu;
  const int     goff  = (wave < 2) ? 0   : H_DIM;
  const int     gtb   = (wave & 1) * 8;          // tile base within gate

  for (int t = 0; t < S; ++t) {
    // ---- load x_t into comb[:,0:H), att/mask for this step ----
    for (int i = tid; i < TB * H_DIM; i += 128) {
      int m = i / H_DIM, c = i % H_DIM;
      comb[m][c] = (__bf16)hist[((size_t)(bBase + m) * S + t) * H_DIM + c];
    }
    if (tid < TB) {
      att_t[tid]  = att [(size_t)(bBase + tid) * S + t];
      mask_t[tid] = mask[(size_t)(bBase + tid) * S + t];
    }
    __syncthreads();

    // ---- gates GEMM: wave computes 8 n-tiles of its gate, as 4 pairs ----
    #pragma clang loop unroll(disable)
    for (int i = 0; i < 4; ++i) {
      int ntl = gtb + 2 * i;                 // tile pair (ntl, ntl+1)
      v8f acc0 = {}, acc1 = {};
      gemm_tile2(&comb[0][0], WPg, H_DIM / 16, ntl, K_DIM / 32, acc0, acc1);
      int nG0 = ntl * 16 + nIdx;             // 0..255 within gate
      int nG1 = nG0 + 16;
      float b0 = biasg[nG0], b1 = biasg[nG1];
      #pragma unroll
      for (int v = 0; v < 8; ++v) {
        int m = grp * 8 + v;
        gbuf[m][goff + nG0] = (__bf16)fast_sigmoid(acc0[v] + b0);
        gbuf[m][goff + nG1] = (__bf16)fast_sigmoid(acc1[v] + b1);
      }
    }
    __syncthreads();

    // ---- comb h-half := bf16(r * h) ----
    for (int i = tid; i < TB * H_DIM; i += 128) {
      int m = i / H_DIM, c = i % H_DIM;
      comb[m][H_DIM + c] = (__bf16)((float)gbuf[m][c] * hbuf[m][c]);
    }
    __syncthreads();

    // ---- candidate GEMM: wave computes 4 n-tiles, as 2 pairs; h update ----
    #pragma clang loop unroll(disable)
    for (int i = 0; i < 2; ++i) {
      int nt = wave * 4 + 2 * i;
      v8f acc0 = {}, acc1 = {};
      gemm_tile2(&comb[0][0], WcP, H_DIM / 16, nt, K_DIM / 32, acc0, acc1);
      int nG0 = nt * 16 + nIdx;
      int nG1 = nG0 + 16;
      float b0 = bc[nG0], b1 = bc[nG1];
      #pragma unroll
      for (int v = 0; v < 8; ++v) {
        int m = grp * 8 + v;
        float a_m = att_t[m];
        float c0  = fast_tanh(acc0[v] + b0);
        float u0  = a_m * (float)gbuf[m][H_DIM + nG0];
        hbuf[m][nG0] = (1.f - u0) * hbuf[m][nG0] + u0 * c0;
        float c1  = fast_tanh(acc1[v] + b1);
        float u1  = a_m * (float)gbuf[m][H_DIM + nG1];
        hbuf[m][nG1] = (1.f - u1) * hbuf[m][nG1] + u1 * c1;
      }
    }
    __syncthreads();

    // ---- comb h-half := bf16(h_new); masked running max ----
    for (int i = tid; i < TB * H_DIM; i += 128) {
      int m = i / H_DIM, c = i % H_DIM;
      float hn = hbuf[m][c];
      comb[m][H_DIM + c] = (__bf16)hn;
      float hm = mask_t[m] ? hn : 0.f;
      float pm = (float)mxh[m][c];
      mxh[m][c] = (__bf16)fmaxf(pm, hm);
    }
    __syncthreads();
  }

  // ---- emit last = h_S ; stage A = bf16(maxh) for pooled GEMM ----
  for (int i = tid; i < TB * H_DIM; i += 128) {
    int m = i / H_DIM, c = i % H_DIM;
    out[(size_t)(bBase + m) * (2 * H_DIM) + c] = hbuf[m][c];
    comb[m][c] = mxh[m][c];
  }
  __syncthreads();

  // ---- pooled GEMM: maxh(16x256) @ Wo^T(256x256) + bo ----
  #pragma clang loop unroll(disable)
  for (int i = 0; i < 2; ++i) {
    int nt = wave * 4 + 2 * i;
    v8f acc0 = {}, acc1 = {};
    gemm_tile2(&comb[0][0], WoP, H_DIM / 16, nt, H_DIM / 32, acc0, acc1);
    int nG0 = nt * 16 + nIdx;
    int nG1 = nG0 + 16;
    float b0 = bo[nG0], b1 = bo[nG1];
    #pragma unroll
    for (int v = 0; v < 8; ++v) {
      int m = grp * 8 + v;
      out[(size_t)(bBase + m) * (2 * H_DIM) + H_DIM + nG0] = acc0[v] + b0;
      out[(size_t)(bBase + m) * (2 * H_DIM) + H_DIM + nG1] = acc1[v] + b1;
    }
  }
}

// ---------------------------------------------------------------------------
// Launcher
// ---------------------------------------------------------------------------
extern "C" void kernel_launch(void* const* d_in, const int* in_sizes, int n_in,
                              void* d_out, int out_size, void* d_ws, size_t ws_size,
                              hipStream_t stream) {
  const float* hist   = (const float*)d_in[0];
  const float* target = (const float*)d_in[1];
  const int*   mask   = (const int*)  d_in[2];
  const float* Wr     = (const float*)d_in[3];
  const float* br     = (const float*)d_in[4];
  const float* Wu     = (const float*)d_in[5];
  const float* bu     = (const float*)d_in[6];
  const float* Wc     = (const float*)d_in[7];
  const float* bc     = (const float*)d_in[8];
  const float* Wo     = (const float*)d_in[9];
  const float* bo     = (const float*)d_in[10];
  float* out = (float*)d_out;

  const int BS_H = in_sizes[0];            // B*S*H
  const int B_H  = in_sizes[1];            // B*H
  const int BS   = in_sizes[2];            // B*S
  const int H = BS_H / BS;                 // 256
  const int B = B_H / H;                   // 1024
  const int S = BS / B;                    // 200

  // workspace layout (aligned to 256B chunks)
  char* ws = (char*)d_ws;
  auto align256 = [](size_t x) { return (x + 255) & ~(size_t)255; };
  float* att = (float*)ws;                 ws += align256((size_t)B * S * sizeof(float));
  __bf16* WrP = (__bf16*)ws;               ws += align256((size_t)512 * 256 * sizeof(__bf16));
  __bf16* WuP = (__bf16*)ws;               ws += align256((size_t)512 * 256 * sizeof(__bf16));
  __bf16* WcP = (__bf16*)ws;               ws += align256((size_t)512 * 256 * sizeof(__bf16));
  __bf16* WoP = (__bf16*)ws;               ws += align256((size_t)256 * 256 * sizeof(__bf16));

  // attention (memory bound, streams hist once)
  attn_kernel<<<B, 256, 0, stream>>>(hist, target, mask, att, S);

  // prepack weights to WMMA B-fragment layout (bf16)
  {
    int thr512 = (512 / 32) * (256 / 16) * 32;   // 8192
    int thr256 = (256 / 32) * (256 / 16) * 32;   // 4096
    prepack_kernel<<<(thr512 + 255) / 256, 256, 0, stream>>>(Wr, WrP, 256, 512);
    prepack_kernel<<<(thr512 + 255) / 256, 256, 0, stream>>>(Wu, WuP, 256, 512);
    prepack_kernel<<<(thr512 + 255) / 256, 256, 0, stream>>>(Wc, WcP, 256, 512);
    prepack_kernel<<<(thr256 + 255) / 256, 256, 0, stream>>>(Wo, WoP, 256, 256);
  }

  // fused scan: one workgroup per 16 batch rows, all S steps
  gru_fused_kernel<<<B / TB, 128, 0, stream>>>(
      hist, mask, att, br, bu, bc, bo, WrP, WuP, WcP, WoP, out, S);
}
